// GCNCoraPerturb_50388556316834
// MI455X (gfx1250) — matrix operand order repaired and checked by reading, not compiled, not run on hardware
//
#include <hip/hip_runtime.h>
#include <math.h>

// ---------------- problem constants ----------------
static constexpr int N   = 8192;   // nodes
static constexpr int NF  = 1433;   // input features
static constexpr int NH  = 64;     // hidden
static constexpr int NC  = 7;      // classes
static constexpr int NCP = 16;     // classes padded to one WMMA tile
static constexpr int NS  = 8;      // k-split factor for adjacency GEMMs

// ---------------- bf16 helpers (RNE) ----------------
__device__ __forceinline__ unsigned short f2bf(float f) {
    unsigned u = __float_as_uint(f);
    u += 0x7FFFu + ((u >> 16) & 1u);
    return (unsigned short)(u >> 16);
}
__device__ __forceinline__ float bf2f(unsigned short h) {
    return __uint_as_float(((unsigned)h) << 16);
}

// ---------------- WMMA types ----------------
typedef __attribute__((ext_vector_type(16))) __bf16 v16bf;
typedef __attribute__((ext_vector_type(8)))  float  v8f;

union ABFrag { v16bf v; uint4 q[2]; };

// =====================================================================
// Kernel 1: build A (bf16) from sub_adj & P_vec (lower triangle only,
// mirrored through LDS), accumulate row sums d via atomics.
// Tile = 64x64, block = 256 threads, grid = (128,128) with tj<=ti live.
// =====================================================================
__global__ void build_A_kernel(const float* __restrict__ sub_adj,
                               const float* __restrict__ P_vec,
                               unsigned short* __restrict__ A,
                               float* __restrict__ d) {
    const int ti = blockIdx.x, tj = blockIdx.y;
    if (tj > ti) return;

    __shared__ unsigned short tile[64][66];
    __shared__ float rowSum[64];
    __shared__ float colSum[64];

    const int t  = threadIdx.x;
    const int tx = t & 63;       // column within tile
    const int tq = t >> 6;       // 0..3

    if (t < 64) { rowSum[t] = 0.f; colSum[t] = 0.f; }

    // phase 1: compute lower-triangle (global) elements of this tile
    #pragma unroll
    for (int m = 0; m < 16; ++m) {
        const int r = tq * 16 + m;
        const int i = ti * 64 + r;
        const int j = tj * 64 + tx;
        const bool live = (ti != tj) || (tx <= r);
        if (live) {
            const float adj = sub_adj[(size_t)i * N + j];
            const size_t pidx = ((size_t)i * (size_t)(i + 1)) / 2 + (size_t)j;
            const float p  = P_vec[pidx];
            const float sg = 1.0f / (1.0f + __expf(-p));
            const float a  = sg * adj + ((i == j) ? 1.0f : 0.0f);
            tile[r][tx] = f2bf(a);
        }
    }
    __syncthreads();

    // diagonal tile: mirror lower into upper (A is symmetric)
    if (ti == tj) {
        #pragma unroll
        for (int m = 0; m < 16; ++m) {
            const int r = tq * 16 + m;
            if (tx > r) tile[r][tx] = tile[tx][r];
        }
        __syncthreads();
    }

    // phase 2: row/col partial sums for degree vector
    #pragma unroll
    for (int m = 0; m < 16; ++m) {
        const int r = tq * 16 + m;
        const float v = bf2f(tile[r][tx]);
        atomicAdd(&rowSum[r], v);
        if (ti != tj) atomicAdd(&colSum[tx], v);
    }
    __syncthreads();

    // phase 3: write tile (and its transpose for off-diagonal tiles)
    #pragma unroll
    for (int m = 0; m < 16; ++m) {
        const int r = tq * 16 + m;
        A[(size_t)(ti * 64 + r) * N + (tj * 64 + tx)] = tile[r][tx];
    }
    if (ti != tj) {
        #pragma unroll
        for (int m = 0; m < 16; ++m) {
            const int r = tq * 16 + m;
            A[(size_t)(tj * 64 + r) * N + (ti * 64 + tx)] = tile[tx][r];
        }
    }

    if (t < 64) {
        atomicAdd(&d[ti * 64 + t], rowSum[t]);
        if (ti != tj) atomicAdd(&d[tj * 64 + t], colSum[t]);
    }
}

// =====================================================================
// Kernel 2: dinv[i] = d[i] > 0 ? rsqrt(d[i]) : 0
// =====================================================================
__global__ void dinv_kernel(const float* __restrict__ d, float* __restrict__ dinv) {
    const int i = blockIdx.x * blockDim.x + threadIdx.x;
    if (i < N) {
        const float v = d[i];
        dinv[i] = (v > 0.f) ? rsqrtf(v) : 0.f;
    }
}

// =====================================================================
// Kernel 3: Y1t[f][j] = dinv[j] * (x @ W1)[j][f]   (bf16, transposed)
// 4 rows per 256-thread block; W1 reads coalesced across f.
// =====================================================================
__global__ void xw1_kernel(const float* __restrict__ x,
                           const float* __restrict__ W1,
                           const float* __restrict__ dinv,
                           unsigned short* __restrict__ Y1t) {
    const int t   = threadIdx.x;
    const int f   = t & 63;
    const int row = blockIdx.x * 4 + (t >> 6);
    float acc = 0.f;
    const float* xr = x + (size_t)row * NF;
    for (int k = 0; k < NF; ++k) {
        acc = fmaf(xr[k], W1[(size_t)k * NH + f], acc);
    }
    Y1t[(size_t)f * N + row] = f2bf(acc * dinv[row]);
}

// =====================================================================
// Kernel 4: WMMA GEMM  partial[s] = A(16 rows, k-slice) @ Bt(NCT*16 cols)
// One wave per (row-tile, k-split). A is bf16 row-major, Bt is bf16
// column-major (Bt[col][k]).  Software-pipelined (2-stage) so loads for
// the next k-step are in flight while the current WMMAs execute; the A
// stream (HBM-resident) is additionally prefetched ahead.
// =====================================================================
template <int NCT>
__global__ void wmma_adj_gemm(const unsigned short* __restrict__ A,
                              const unsigned short* __restrict__ Bt,
                              float* __restrict__ partial,
                              int kPerSplit) {
    const int lane = threadIdx.x;          // 0..31
    const int r    = lane & 15;
    const int half = lane >> 4;            // 0 or 1
    const int rowBase = blockIdx.x * 16;
    const int s       = blockIdx.y;
    const int k0      = s * kPerSplit;
    const int kend    = k0 + kPerSplit;    // kPerSplit is a multiple of 64

    // A fragment: lanes 0-15 hold K {0..7,16..23}, lanes 16-31 K {8..15,24..31}
    const unsigned short* arow = A + (size_t)(rowBase + r) * N + half * 8;
    // B fragment: lanes 0-15 hold K {0..7,16..23} of col, lanes 16-31 K {8..15,24..31}
    const unsigned short* bcol[NCT];
    #pragma unroll
    for (int ct = 0; ct < NCT; ++ct)
        bcol[ct] = Bt + (size_t)(ct * 16 + r) * N + half * 8;

    auto loadA = [&](int kk, ABFrag& a) {
        a.q[0] = *(const uint4*)(arow + kk);
        a.q[1] = *(const uint4*)(arow + kk + 16);
    };
    auto loadB = [&](int kk, ABFrag* b) {
        #pragma unroll
        for (int ct = 0; ct < NCT; ++ct) {
            const unsigned short* bp = bcol[ct] + kk;
            b[ct].q[0] = *(const uint4*)(bp);
            b[ct].q[1] = *(const uint4*)(bp + 16);
        }
    };

    v8f acc[NCT];
    #pragma unroll
    for (int ct = 0; ct < NCT; ++ct) acc[ct] = (v8f)(0.0f);

    ABFrag a0, a1, b0[NCT], b1[NCT];
    int kk = k0;
    loadA(kk, a0);
    loadB(kk, b0);

    for (; kk + 64 < kend; kk += 64) {
        // keep the HBM A-stream ahead of demand
        __builtin_prefetch(arow + kk + 256, 0, 0);
        loadA(kk + 32, a1);
        loadB(kk + 32, b1);
        #pragma unroll
        for (int ct = 0; ct < NCT; ++ct)
            acc[ct] = __builtin_amdgcn_wmma_f32_16x16x32_bf16(
                false, a0.v, false, b0[ct].v, (short)0, acc[ct], false, false);
        loadA(kk + 64, a0);
        loadB(kk + 64, b0);
        #pragma unroll
        for (int ct = 0; ct < NCT; ++ct)
            acc[ct] = __builtin_amdgcn_wmma_f32_16x16x32_bf16(
                false, a1.v, false, b1[ct].v, (short)0, acc[ct], false, false);
    }
    // tail: exactly two k-steps remain (kk+64 == kend)
    loadA(kk + 32, a1);
    loadB(kk + 32, b1);
    #pragma unroll
    for (int ct = 0; ct < NCT; ++ct)
        acc[ct] = __builtin_amdgcn_wmma_f32_16x16x32_bf16(
            false, a0.v, false, b0[ct].v, (short)0, acc[ct], false, false);
    #pragma unroll
    for (int ct = 0; ct < NCT; ++ct)
        acc[ct] = __builtin_amdgcn_wmma_f32_16x16x32_bf16(
            false, a1.v, false, b1[ct].v, (short)0, acc[ct], false, false);

    // C/D layout: VGPR v -> row v + 8*half, col = lane&15
    #pragma unroll
    for (int ct = 0; ct < NCT; ++ct) {
        #pragma unroll
        for (int v = 0; v < 8; ++v) {
            const int rl = v + half * 8;
            partial[((size_t)s * N + rowBase + rl) * (NCT * 16) + ct * 16 + r] = acc[ct][v];
        }
    }
}

// =====================================================================
// Kernel 5: combine layer-1 partials -> h1 = relu(dinv*sum + b1),
// then fuse h1 @ W2 and write Y2t[c][j] = dinv[j] * (h1@W2)[j][c] (bf16)
// =====================================================================
__global__ void combine1_kernel(const float* __restrict__ P1,
                                const float* __restrict__ dinv,
                                const float* __restrict__ b1,
                                const float* __restrict__ W2,
                                unsigned short* __restrict__ Y2t) {
    const int row = blockIdx.x * blockDim.x + threadIdx.x;
    if (row >= N) return;

    float h[NH];
    #pragma unroll
    for (int f = 0; f < NH; ++f) h[f] = 0.f;
    #pragma unroll
    for (int s = 0; s < NS; ++s) {
        const float4* p = (const float4*)(P1 + ((size_t)s * N + row) * NH);
        #pragma unroll
        for (int q = 0; q < NH / 4; ++q) {
            const float4 v = p[q];
            h[4 * q + 0] += v.x; h[4 * q + 1] += v.y;
            h[4 * q + 2] += v.z; h[4 * q + 3] += v.w;
        }
    }
    const float di = dinv[row];
    #pragma unroll
    for (int f = 0; f < NH; ++f) {
        const float v = di * h[f] + b1[f];
        h[f] = v > 0.f ? v : 0.f;
    }
    #pragma unroll
    for (int c = 0; c < NC; ++c) {
        float acc = 0.f;
        #pragma unroll
        for (int f = 0; f < NH; ++f) acc = fmaf(h[f], W2[(size_t)f * NC + c], acc);
        Y2t[(size_t)c * N + row] = f2bf(di * acc);
    }
    #pragma unroll
    for (int c = NC; c < NCP; ++c) Y2t[(size_t)c * N + row] = 0;  // pad cols = 0
}

// =====================================================================
// Kernel 6: combine layer-2 partials, add b2, log_softmax over 7 classes
// =====================================================================
__global__ void final_kernel(const float* __restrict__ P2,
                             const float* __restrict__ dinv,
                             const float* __restrict__ b2,
                             float* __restrict__ out) {
    const int row = blockIdx.x * blockDim.x + threadIdx.x;
    if (row >= N) return;
    const float di = dinv[row];
    float sarr[NC];
    #pragma unroll
    for (int c = 0; c < NC; ++c) {
        float acc = 0.f;
        #pragma unroll
        for (int s = 0; s < NS; ++s) acc += P2[((size_t)s * N + row) * NCP + c];
        sarr[c] = di * acc + b2[c];
    }
    float m = sarr[0];
    #pragma unroll
    for (int c = 1; c < NC; ++c) m = fmaxf(m, sarr[c]);
    float z = 0.f;
    #pragma unroll
    for (int c = 0; c < NC; ++c) z += expf(sarr[c] - m);
    const float lse = logf(z);
    #pragma unroll
    for (int c = 0; c < NC; ++c) out[(size_t)row * NC + c] = sarr[c] - m - lse;
}

// =====================================================================
// Host-side launcher
// =====================================================================
extern "C" void kernel_launch(void* const* d_in, const int* in_sizes, int n_in,
                              void* d_out, int out_size, void* d_ws, size_t ws_size,
                              hipStream_t stream) {
    const float* x       = (const float*)d_in[0];  // N x NF
    const float* sub_adj = (const float*)d_in[1];  // N x N
    const float* P_vec   = (const float*)d_in[2];  // N(N+1)/2
    const float* W1      = (const float*)d_in[3];  // NF x NH
    const float* b1      = (const float*)d_in[4];  // NH
    const float* W2      = (const float*)d_in[5];  // NH x NC
    const float* b2      = (const float*)d_in[6];  // NC
    float* out = (float*)d_out;                    // N x NC

    // workspace layout (256B aligned chunks)
    char* ws = (char*)d_ws;
    size_t off = 0;
    auto alloc = [&](size_t bytes) { void* p = ws + off; off = (off + bytes + 255) & ~(size_t)255; return p; };
    unsigned short* A    = (unsigned short*)alloc((size_t)N * N * 2);      // 128 MB bf16 adjacency
    float*          dsum = (float*)alloc((size_t)N * 4);
    float*          dinv = (float*)alloc((size_t)N * 4);
    unsigned short* Y1t  = (unsigned short*)alloc((size_t)NH * N * 2);     // 1 MB  (col-major)
    unsigned short* Y2t  = (unsigned short*)alloc((size_t)NCP * N * 2);    // 256 KB (col-major)
    float*          P1   = (float*)alloc((size_t)NS * N * NH * 4);         // 16 MB layer-1 k-split partials
    float*          P2   = (float*)alloc((size_t)NS * N * NCP * 4);        // 4 MB  layer-2 k-split partials

    // 1) degree accumulator must start at zero
    hipMemsetAsync(dsum, 0, (size_t)N * 4, stream);

    // 2) build bf16 A + row sums (lower triangle pass, mirrored)
    build_A_kernel<<<dim3(N / 64, N / 64), 256, 0, stream>>>(sub_adj, P_vec, A, dsum);

    // 3) dinv = rsqrt(d)
    dinv_kernel<<<N / 256, 256, 0, stream>>>(dsum, dinv);

    // 4) Y1t = dinv ⊗ (x @ W1), transposed bf16
    xw1_kernel<<<N / 4, 256, 0, stream>>>(x, W1, dinv, Y1t);

    // 5) layer 1: P1[s] = A @ Y1  (wave-per-tile WMMA, NS-way k-split)
    wmma_adj_gemm<4><<<dim3(N / 16, NS), 32, 0, stream>>>(A, Y1t, P1, N / NS);

    // 6) h1 = relu(dinv*sum + b1); Y2t = dinv ⊗ (h1 @ W2), transposed bf16
    combine1_kernel<<<N / 256, 256, 0, stream>>>(P1, dinv, b1, W2, Y2t);

    // 7) layer 2: P2[s] = A @ Y2
    wmma_adj_gemm<1><<<dim3(N / 16, NS), 32, 0, stream>>>(A, Y2t, P2, N / NS);

    // 8) bias + log_softmax
    final_kernel<<<N / 256, 256, 0, stream>>>(P2, dinv, b2, out);

    (void)in_sizes; (void)n_in; (void)out_size; (void)ws_size;
}
